// ConvGenQuantum_65481071403142
// MI455X (gfx1250) — compile-verified, matrix operand order checked by reading, not run on hardware
//
#include <hip/hip_runtime.h>

typedef __attribute__((ext_vector_type(16))) _Float16 v16h;
typedef __attribute__((ext_vector_type(8)))  _Float16 v8h;
typedef __attribute__((ext_vector_type(8)))  float    v8f;
typedef __attribute__((ext_vector_type(2)))  float    v2f;

#if __has_builtin(__builtin_amdgcn_wmma_f32_16x16x4_f32)
#define USE_F32_WMMA 1
#else
#define USE_F32_WMMA 0
#endif

struct c2c { float x, y; };
__device__ __forceinline__ c2c cmul(c2c a, c2c b) { return { a.x*b.x - a.y*b.y, a.x*b.y + a.y*b.x }; }
__device__ __forceinline__ c2c cadd(c2c a, c2c b) { return { a.x+b.x, a.y+b.y }; }

// ---------------------------------------------------------------------------
// Setup: build A_w = Re(R^dagger Z_w R) (4 real symmetric 16x16) from theta[5].
// One block, 256 threads; thread t owns entry (i=t>>4, j=t&15) of R in LDS.
// Gate order matches RAND_OPS.  Wire w -> bit position (3-w).
// Emits the table both as f32 (for the exact f32-WMMA path) and f16 fallback.
// ---------------------------------------------------------------------------
__global__ void quanv_setup(const float* __restrict__ theta,
                            float* __restrict__ Af32,      /* [4][16][16] f32 */
                            _Float16* __restrict__ Af16)   /* [4][16][16] f16 */
{
  __shared__ c2c R[256];
  const int t = threadIdx.x;
  const int i = t >> 4, j = t & 15;
  R[t] = { (i == j) ? 1.0f : 0.0f, 0.0f };
  __syncthreads();

  auto apply1q = [&](c2c g00, c2c g01, c2c g10, c2c g11, int wire) {
    int bp = 3 - wire;
    int i0 = i & ~(1 << bp), i1 = i | (1 << bp);
    c2c v0 = R[i0 * 16 + j], v1 = R[i1 * 16 + j];
    __syncthreads();
    R[t] = ((i >> bp) & 1) ? cadd(cmul(g10, v0), cmul(g11, v1))
                           : cadd(cmul(g00, v0), cmul(g01, v1));
    __syncthreads();
  };
  auto applycx = [&](int cw, int tw) {
    int cb = 3 - cw, tb = 3 - tw;
    int src = ((i >> cb) & 1) ? (i ^ (1 << tb)) : i;
    c2c v = R[src * 16 + j];
    __syncthreads();
    R[t] = v;
    __syncthreads();
  };

  float t0 = theta[0], t1 = theta[1], t2 = theta[2], t3 = theta[3], t4 = theta[4];
  { float ch = cosf(t0*0.5f), sh = sinf(t0*0.5f);                 // ry w0
    apply1q({ch,0},{-sh,0},{sh,0},{ch,0}, 0); }
  { float ch = cosf(t1*0.5f), sh = sinf(t1*0.5f);                 // rx w1
    apply1q({ch,0},{0,-sh},{0,-sh},{ch,0}, 1); }
  applycx(0, 1);
  { float ch = cosf(t2*0.5f), sh = sinf(t2*0.5f);                 // rz w2
    apply1q({ch,-sh},{0,0},{0,0},{ch,sh}, 2); }
  applycx(1, 2);
  { float ch = cosf(t3*0.5f), sh = sinf(t3*0.5f);                 // ry w3
    apply1q({ch,0},{-sh,0},{sh,0},{ch,0}, 3); }
  applycx(2, 3);
  { float ch = cosf(t4*0.5f), sh = sinf(t4*0.5f);                 // rx w0
    apply1q({ch,0},{0,-sh},{0,-sh},{ch,0}, 0); }

  // A_w[i][j] = sum_k z_w(k) * Re( conj(R[k][i]) * R[k][j] )
  float acc0 = 0.f, acc1 = 0.f, acc2 = 0.f, acc3 = 0.f;
  #pragma unroll
  for (int k = 0; k < 16; ++k) {
    c2c ri = R[k * 16 + i], rj = R[k * 16 + j];
    float re = ri.x * rj.x + ri.y * rj.y;
    acc0 += ((k >> 3) & 1) ? -re : re;
    acc1 += ((k >> 2) & 1) ? -re : re;
    acc2 += ((k >> 1) & 1) ? -re : re;
    acc3 += ((k >> 0) & 1) ? -re : re;
  }
  Af32[0 * 256 + t] = acc0;  Af16[0 * 256 + t] = (_Float16)acc0;
  Af32[1 * 256 + t] = acc1;  Af16[1 * 256 + t] = (_Float16)acc1;
  Af32[2 * 256 + t] = acc2;  Af16[2 * 256 + t] = (_Float16)acc2;
  Af32[3 * 256 + t] = acc3;  Af16[3 * 256 + t] = (_Float16)acc3;
}

// branchless 8-way select from a v8f accumulator (avoid dynamic vector index)
__device__ __forceinline__ float sel8(v8f c, int k) {
  float r = c[0];
  r = (k == 1) ? c[1] : r;  r = (k == 2) ? c[2] : r;
  r = (k == 3) ? c[3] : r;  r = (k == 4) ? c[4] : r;
  r = (k == 5) ? c[5] : r;  r = (k == 6) ? c[6] : r;
  r = (k == 7) ? c[7] : r;
  return r;
}

// ---------------------------------------------------------------------------
// Main: each wave32 handles 16 patches.  Per patch build the real product
// state psi[16]; per wire w compute diag(Psi * A_w * Psi^T) on the matrix
// pipes.  Block = 256 threads = 8 waves = 128 patches.
//
// Preferred path: exact fp32 via chained V_WMMA_F32_16X16X4_F32 (K=4 x 4).
// Layout identity used: for the 16x4 f32 A-layout (lanes 0-15 hold K=0,1;
// lanes 16-31 hold K=2,3) and mirrored 4x16 B-layout, the per-lane registers
// of A-frag(Psi chunk) and B-frag(Psi^T chunk) coincide -> no shuffle for
// the first product.
// ---------------------------------------------------------------------------
__global__ void __launch_bounds__(256)
quanv_main(const float* __restrict__ x, const float* __restrict__ Af32,
           const _Float16* __restrict__ Af16, float* __restrict__ out, int NP)
{
  const int lane = threadIdx.x & 31;
  const int wave = threadIdx.x >> 5;
  const int m    = lane & 15;   // patch slot within wave (WMMA M / N index)
  const int hi   = lane >> 4;   // lane half selector
  const bool lo16 = (hi == 0);

  const int P  = blockIdx.x * 128 + wave * 16 + m;
  const int Pc = (P < NP) ? P : (NP - 1);

  // patch -> image coordinates (B,196) -> (r,c) in 14x14 grid of 2x2 blocks
  const int b  = Pc / 196;
  const int rc = Pc - b * 196;
  const int r  = rc / 14;
  const int c  = rc - r * 14;
  const float* img = x + b * 784 + (2 * r) * 28 + 2 * c;
  const float p0 = img[0], p1 = img[1], p2 = img[28], p3 = img[29];

  // encoder: RY(p)|0> = [cos(p/2), sin(p/2)] per wire
  const float c0 = cosf(p0 * 0.5f), s0 = sinf(p0 * 0.5f);
  const float c1 = cosf(p1 * 0.5f), s1 = sinf(p1 * 0.5f);
  const float c2 = cosf(p2 * 0.5f), s2 = sinf(p2 * 0.5f);
  const float c3 = cosf(p3 * 0.5f), s3 = sinf(p3 * 0.5f);

  const v8f cz = {0.f, 0.f, 0.f, 0.f, 0.f, 0.f, 0.f, 0.f};
  float ew[4];

#if USE_F32_WMMA
  // psi index i = (q<<2) | (hi<<1) | r  -> bits (w0,w1,w2,w3) = (q>>1, q&1, hi, r)
  const float f2 = hi ? s2 : c2;
  v2f aPsi[4];
  #pragma unroll
  for (int q = 0; q < 4; ++q) {
    const float f0 = (q >> 1) ? s0 : c0;
    const float f1 = (q & 1)  ? s1 : c1;
    const float g  = f0 * f1 * f2;
    v2f a = { g * c3, g * s3 };
    aPsi[q] = a;
  }

  #pragma unroll
  for (int w = 0; w < 4; ++w) {
    // A_w chunk q A-frag: lane row = m, K pair = 4q + 2*hi + {0,1}
    const float* Arow = Af32 + w * 256 + m * 16 + 2 * hi;

    // T_w[i][n] = sum_k A_w[i,k] psi_n[k]   (B-frag(Psi^T chunk) == aPsi[q])
    v8f T = cz;
    #pragma unroll
    for (int q = 0; q < 4; ++q) {
      const v2f a = *(const v2f*)(Arow + 4 * q);
      T = __builtin_amdgcn_wmma_f32_16x16x4_f32(
          false, a, false, aPsi[q], (short)0, T, false, false);
    }

    // Repack T (C-layout: lane holds col n=lane&15, rows v+8*hi) into B-frags.
    float own[8], oth[8];
    #pragma unroll
    for (int e = 0; e < 8; ++e) { own[e] = T[e]; oth[e] = __shfl_xor(own[e], 16); }
    // lane hi=0 holds rows 0..7 of col n, partner holds rows 8..15 (and vice versa)
    v2f bq[4];
    { v2f lo = {own[0], own[1]}, hi_ = {oth[2], oth[3]}; bq[0] = lo16 ? lo : hi_; }
    { v2f lo = {own[4], own[5]}, hi_ = {oth[6], oth[7]}; bq[1] = lo16 ? lo : hi_; }
    { v2f lo = {oth[0], oth[1]}, hi_ = {own[2], own[3]}; bq[2] = lo16 ? lo : hi_; }
    { v2f lo = {oth[4], oth[5]}, hi_ = {own[6], own[7]}; bq[3] = lo16 ? lo : hi_; }

    // E[p][n] = sum_k psi_p[k] T[k][n]; diagonal = quadratic forms
    v8f E = cz;
    #pragma unroll
    for (int q = 0; q < 4; ++q) {
      E = __builtin_amdgcn_wmma_f32_16x16x4_f32(
          false, aPsi[q], false, bq[q], (short)0, E, false, false);
    }
    ew[w] = sel8(E, lane & 7);
  }
#else
  // -------- f16 fallback path (codegen-confirmed wmma_f32_16x16x32_f16) ----
  const float f0 = lo16 ? c0 : s0;
  float psi[8];
  #pragma unroll
  for (int k = 0; k < 8; ++k) {
    const float f1 = ((k >> 2) & 1) ? s1 : c1;
    const float f2 = ((k >> 1) & 1) ? s2 : c2;
    const float f3 = ((k >> 0) & 1) ? s3 : c3;
    psi[k] = f0 * f1 * f2 * f3;
  }
  const _Float16 hz = (_Float16)0.0f;
  v16h aPsi;
  #pragma unroll
  for (int k = 0; k < 8; ++k) { aPsi[k] = (_Float16)psi[k]; aPsi[8 + k] = hz; }

  union HU { _Float16 h[2]; unsigned u; };
  unsigned pk[4], ok[4];
  #pragma unroll
  for (int n = 0; n < 4; ++n) {
    HU u; u.h[0] = (_Float16)psi[2 * n]; u.h[1] = (_Float16)psi[2 * n + 1];
    pk[n] = u.u;
    ok[n] = __shfl_xor(pk[n], 16);
  }
  v16h bPsiT;
  #pragma unroll
  for (int n = 0; n < 4; ++n) {
    HU a; a.u = pk[n];
    HU o; o.u = ok[n];
    bPsiT[2 * n]         = lo16 ? a.h[0] : hz;
    bPsiT[2 * n + 1]     = lo16 ? a.h[1] : hz;
    bPsiT[8 + 2 * n]     = lo16 ? o.h[0] : hz;
    bPsiT[8 + 2 * n + 1] = lo16 ? o.h[1] : hz;
  }

  #pragma unroll
  for (int w = 0; w < 4; ++w) {
    const v8h arow = *(const v8h*)(Af16 + w * 256 + m * 16 + hi * 8);
    v16h aA;
    #pragma unroll
    for (int k = 0; k < 8; ++k) { aA[k] = arow[k]; aA[8 + k] = hz; }

    v8f cT = __builtin_amdgcn_wmma_f32_16x16x32_f16(
        false, aA, false, bPsiT, (short)0, cz, false, false);

    v16h bT;
    #pragma unroll
    for (int e = 0; e < 8; ++e) {
      const float own = cT[e];
      const float oth = __shfl_xor(own, 16);
      bT[e]     = lo16 ? (_Float16)own : hz;
      bT[8 + e] = lo16 ? (_Float16)oth : hz;
    }

    v8f cE = __builtin_amdgcn_wmma_f32_16x16x32_f16(
        false, aPsi, false, bT, (short)0, cz, false, false);
    ew[w] = sel8(cE, lane & 7);
  }
#endif

  // lanes {0..7, 24..31} carry the 16 diagonal entries; their patch index is
  // exactly P (lane&15 == diag row).  Packed 16B store per patch.
  if (((lane < 8) || (lane >= 24)) && P < NP) {
    float4 o; o.x = ew[0]; o.y = ew[1]; o.z = ew[2]; o.w = ew[3];
    *reinterpret_cast<float4*>(out + (size_t)P * 4) = o;
  }
}

extern "C" void kernel_launch(void* const* d_in, const int* in_sizes, int n_in,
                              void* d_out, int out_size, void* d_ws, size_t ws_size,
                              hipStream_t stream) {
  (void)n_in; (void)out_size; (void)ws_size;
  const float* x     = (const float*)d_in[0];
  const float* theta = (const float*)d_in[1];
  float* out         = (float*)d_out;
  float* Af32        = (float*)d_ws;                         // 4 KB
  _Float16* Af16     = (_Float16*)((char*)d_ws + 4096);      // 2 KB

  const int B  = in_sizes[0] / 784;       // images
  const int NP = B * 196;                 // patches

  quanv_setup<<<1, 256, 0, stream>>>(theta, Af32, Af16);
  const int blocks = (NP + 127) / 128;    // 128 patches per 256-thread block
  quanv_main<<<blocks, 256, 0, stream>>>(x, Af32, Af16, out, NP);
}
